// PGN_72206990181061
// MI455X (gfx1250) — compile-verified, matrix operand order checked by reading, not run on hardware
//
#include <hip/hip_runtime.h>
#include <hip/hip_bf16.h>

typedef __attribute__((ext_vector_type(16))) _Float16 v16h;
typedef __attribute__((ext_vector_type(8)))  _Float16 v8h;
typedef __attribute__((ext_vector_type(8)))  float    v8f;

#define LSTEPS 4
#define BATCH  8
#define KN     256   // nodes per graph
#define ENC    128
#define HID    128
#define QD     64
#define ROWS   (BATCH*KN)   // 2048 node-rows

// ---------------------------------------------------------------------------
// Fast f16 fragment loaders.
// A fragment (16x32 MxK) from row-major f16 [M,lda]: lane&15 = M row; per lane
// the 16 halves are two contiguous 8-half runs: K = k0+8*hlf+{0..7, 16..23}.
// Both runs are 16B aligned by construction -> 2x global_load_b128.
// ---------------------------------------------------------------------------
__device__ __forceinline__ v16h load_a_frag_h(const _Float16* __restrict__ A, int lda,
                                              int m, int k0, int hlf) {
  const _Float16* base = A + m * lda + k0 + (hlf << 3);
  v8h lo = *reinterpret_cast<const v8h*>(base);
  v8h hi = *reinterpret_cast<const v8h*>(base + 16);
  return __builtin_shufflevector(lo, hi, 0,1,2,3,4,5,6,7,8,9,10,11,12,13,14,15);
}

// B fragment from pre-packed fragment-order weights: one aligned 32B load/lane.
__device__ __forceinline__ v16h load_bp_frag(const _Float16* __restrict__ Wp, int tilesn,
                                             int k0, int n0, int lane) {
  return *reinterpret_cast<const v16h*>(
      Wp + ((((k0 >> 5) * tilesn + (n0 >> 4)) << 9) + (lane << 4)));
}

// One wave: 16x16 f32 tile of Ah[M,K](f16,lda) @ Wp(packed), K multiple of 32.
__device__ __forceinline__ v8f wmma_tile_h(const _Float16* __restrict__ Ah, int lda,
                                           const _Float16* __restrict__ Wp, int tilesn,
                                           int m0, int n0, int K, int lane) {
  int hlf = lane >> 4;
  int m = m0 + (lane & 15);
  v8f c = {0.f, 0.f, 0.f, 0.f, 0.f, 0.f, 0.f, 0.f};
  for (int k0 = 0; k0 < K; k0 += 32) {
    v16h a = load_a_frag_h(Ah, lda, m, k0, hlf);
    v16h b = load_bp_frag(Wp, tilesn, k0, n0, lane);
    c = __builtin_amdgcn_wmma_f32_16x16x32_f16(false, a, false, b, (short)0, c,
                                               false, false);
  }
  return c;
}

// ---------------------------------------------------------------------------
// One-time weight pre-pack: f32 row-major [K,N] -> f16 fragment order.
// One thread per (k-chunk, n-tile, lane); writes its 16 halves contiguously.
// ---------------------------------------------------------------------------
__global__ void pack_w_kernel(const float* __restrict__ W, _Float16* __restrict__ Wp,
                              int K, int N) {
  int tid = blockIdx.x * blockDim.x + threadIdx.x;
  int tilesn = N >> 4;
  int total = (K >> 5) * tilesn * 32;
  if (tid >= total) return;
  int lane = tid & 31;
  int t = tid >> 5;
  int n0 = (t % tilesn) << 4;
  int k0 = (t / tilesn) << 5;
  int hlf = lane >> 4;
  int n = n0 + (lane & 15);
  _Float16* out = Wp + (t << 9) + (lane << 4);
#pragma unroll
  for (int v = 0; v < 8; ++v) {
    int kb = k0 + ((v >> 2) << 4) + (hlf << 3) + ((v & 3) << 1);
    out[2 * v]     = (_Float16)W[kb * N + n];
    out[2 * v + 1] = (_Float16)W[(kb + 1) * N + n];
  }
}

// ---------------------------------------------------------------------------
// z = h @ Wenc[2:,:] + E0*Wenc[0,:] + E1*Wenc[1,:] + b (rank-2 epilogue).
// Writes z (f32, for reductions) and zm_h[:, :128] (f16 GEMM operand).
// ---------------------------------------------------------------------------
__global__ void encode_kernel(const float* __restrict__ Et, const _Float16* __restrict__ h16,
                              const _Float16* __restrict__ WpEnc,
                              const float* __restrict__ Wenc, const float* __restrict__ benc,
                              float* __restrict__ z, _Float16* __restrict__ zm_h) {
  int wid  = (blockIdx.x * blockDim.x + threadIdx.x) >> 5;  // 0..1023
  int lane = threadIdx.x & 31;
  int m0 = (wid >> 3) << 4, n0 = (wid & 7) << 4;            // 128 x 8 tiles
  v8f c = wmma_tile_h(h16, HID, WpEnc, ENC >> 4, m0, n0, HID, lane);
  int hlf = lane >> 4;
  int col = n0 + (lane & 15);
  float w0 = Wenc[col], w1 = Wenc[ENC + col], bb = benc[col];
#pragma unroll
  for (int r = 0; r < 8; ++r) {
    int row = m0 + (hlf << 3) + r;
    float val = c[r] + bb + Et[row * 2] * w0 + Et[row * 2 + 1] * w1;
    z[row * ENC + col] = val;
    zm_h[row * (2 * ENC) + col] = (_Float16)val;
  }
}

// Generic GEMM: [relu](Ah[ROWS,K] @ Wp + b) -> optional f32 and/or f16 outputs.
__global__ void gemm_h_kernel(const _Float16* __restrict__ Ah, int lda, int K,
                              const _Float16* __restrict__ Wp,
                              const float* __restrict__ bias,
                              float* __restrict__ Df, _Float16* __restrict__ Dh,
                              int N, int relu) {
  int wid  = (blockIdx.x * blockDim.x + threadIdx.x) >> 5;
  int lane = threadIdx.x & 31;
  int tilesn = N >> 4;
  int m0 = (wid / tilesn) << 4, n0 = (wid % tilesn) << 4;
  v8f c = wmma_tile_h(Ah, lda, Wp, tilesn, m0, n0, K, lane);
  int hlf = lane >> 4;
  int col = n0 + (lane & 15);
  float bb = bias[col];
#pragma unroll
  for (int r = 0; r < 8; ++r) {
    int row = m0 + (hlf << 3) + r;
    float v = c[r] + bb;
    if (relu) v = fmaxf(v, 0.f);
    if (Df) Df[row * N + col] = v;
    if (Dh) Dh[row * N + col] = (_Float16)v;
  }
}

// m[b,i,:] = max_{j : P[b,j,i]} relu(m1z[b,i,:] + m2z[b,j,:]) -> zm_h[:,128:256]
__global__ void message_kernel(const unsigned char* __restrict__ Pt,
                               const float* __restrict__ m1z, const float* __restrict__ m2z,
                               _Float16* __restrict__ zm_h) {
  int wid  = (blockIdx.x * blockDim.x + threadIdx.x) >> 5;  // 0..2047 = (b,i)
  int lane = threadIdx.x & 31;
  int b = wid >> 8, i = wid & (KN - 1);
  const float* m1r = m1z + (b * KN + i) * ENC;
  float mi[4], acc[4];
#pragma unroll
  for (int u = 0; u < 4; ++u) { mi[u] = m1r[lane + 32 * u]; acc[u] = -__builtin_inff(); }
  const unsigned char* Pb = Pt + (size_t)b * KN * KN + i;   // Pb[j*KN] = P[b,j,i]
  const float* m2b = m2z + b * KN * ENC;
  for (int j = 0; j < KN; ++j) {
    if ((j & 31) == 0 && j + 64 < KN)
      __builtin_prefetch(Pb + (size_t)(j + 64) * KN, 0, 1);  // global_prefetch_b8
    if (Pb[(size_t)j * KN]) {
      const float* m2r = m2b + j * ENC;
#pragma unroll
      for (int u = 0; u < 4; ++u)
        acc[u] = fmaxf(acc[u], fmaxf(mi[u] + m2r[lane + 32 * u], 0.f));
    }
  }
  _Float16* out = zm_h + (b * KN + i) * (2 * ENC) + ENC;
#pragma unroll
  for (int u = 0; u < 4; ++u) out[lane + 32 * u] = (_Float16)acc[u];
}

// A[b] = q[b] @ k[b]^T (256x256, K=64). k's B-fragment has the same access
// pattern as an A-fragment (Q-dim contiguous), so reuse the fast loader.
__global__ void attn_kernel(const _Float16* __restrict__ q, const _Float16* __restrict__ kk,
                            float* __restrict__ Aout) {
  int wid  = (blockIdx.x * blockDim.x + threadIdx.x) >> 5;  // 0..2047
  int lane = threadIdx.x & 31;
  int b = wid >> 8, tile = wid & 255;
  int m0 = (tile >> 4) << 4, n0 = (tile & 15) << 4;
  int hlf = lane >> 4;
  int m = m0 + (lane & 15);
  int n = n0 + (lane & 15);
  const _Float16* qb = q + b * KN * QD;
  const _Float16* kb = kk + b * KN * QD;
  v8f c = {0.f, 0.f, 0.f, 0.f, 0.f, 0.f, 0.f, 0.f};
#pragma unroll
  for (int k0 = 0; k0 < QD; k0 += 32) {
    v16h a  = load_a_frag_h(qb, QD, m, k0, hlf);
    v16h bf = load_a_frag_h(kb, QD, n, k0, hlf);
    c = __builtin_amdgcn_wmma_f32_16x16x32_f16(false, a, false, bf, (short)0, c,
                                               false, false);
  }
  int col = n0 + (lane & 15);
#pragma unroll
  for (int r = 0; r < 8; ++r) {
    int row = m0 + (hlf << 3) + r;
    Aout[(b * KN + row) * KN + col] = c[r];
  }
}

// y[b] = sigmoid(dec([max_k z | max_k hn]))
__global__ void reduce_y_kernel(const float* __restrict__ z, const float* __restrict__ hn,
                                const float* __restrict__ wdec, const float* __restrict__ bdec,
                                float* __restrict__ Y) {
  __shared__ float sred[128];
  int b = blockIdx.x, c = threadIdx.x;
  float zmx = -__builtin_inff(), hmx = -__builtin_inff();
  const float* zb = z + b * KN * ENC;
  const float* hb = hn + b * KN * HID;
  for (int k = 0; k < KN; ++k) {
    zmx = fmaxf(zmx, zb[k * ENC + c]);
    hmx = fmaxf(hmx, hb[k * HID + c]);
  }
  sred[c] = zmx * wdec[c] + hmx * wdec[ENC + c];
  __syncthreads();
  for (int s = 64; s > 0; s >>= 1) {
    if (c < s) sred[c] += sred[c + s];
    __syncthreads();
  }
  if (c == 0) Y[b] = 1.f / (1.f + __expf(-(sred[0] + bdec[0])));
}

// M[b,k] = sigmoid(mask([z | hn])) — one wave per node-row
__global__ void mask_kernel(const float* __restrict__ z, const float* __restrict__ hn,
                            const float* __restrict__ wm, const float* __restrict__ bm,
                            float* __restrict__ Mout) {
  int wid  = (blockIdx.x * blockDim.x + threadIdx.x) >> 5;  // row 0..2047
  int lane = threadIdx.x & 31;
  const float* zr = z + wid * ENC;
  const float* hr = hn + wid * HID;
  float acc = 0.f;
#pragma unroll
  for (int u = 0; u < 4; ++u) {
    int cc = lane + 32 * u;
    acc += zr[cc] * wm[cc] + hr[cc] * wm[ENC + cc];
  }
#pragma unroll
  for (int off = 16; off > 0; off >>= 1) acc += __shfl_xor(acc, off, 32);
  if (lane == 0) Mout[wid] = 1.f / (1.f + __expf(-(acc + bm[0])));
}

// ---------------------------------------------------------------------------
extern "C" void kernel_launch(void* const* d_in, const int* in_sizes, int n_in,
                              void* d_out, int out_size, void* d_ws, size_t ws_size,
                              hipStream_t stream) {
  (void)in_sizes; (void)n_in; (void)out_size; (void)ws_size;
  const float* E          = (const float*)d_in[0];
  const unsigned char* P  = (const unsigned char*)d_in[1];   // bool adjacency
  const float* Wenc = (const float*)d_in[2];
  const float* benc = (const float*)d_in[3];
  const float* M1w  = (const float*)d_in[4];
  const float* M1b  = (const float*)d_in[5];
  const float* M2w  = (const float*)d_in[6];
  const float* M2b  = (const float*)d_in[7];
  const float* Uw   = (const float*)d_in[8];
  const float* Ub   = (const float*)d_in[9];
  const float* decw = (const float*)d_in[10];
  const float* decb = (const float*)d_in[11];
  const float* mskw = (const float*)d_in[12];
  const float* mskb = (const float*)d_in[13];
  const float* Wq   = (const float*)d_in[14];
  const float* bq   = (const float*)d_in[15];
  const float* Wk   = (const float*)d_in[16];
  const float* bk   = (const float*)d_in[17];

  // ---- workspace carve-up (~7 MB, L2 resident) ----
  float* z   = (float*)d_ws;            // [2048,128] f32
  float* m1z = z   + ROWS * ENC;
  float* m2z = m1z + ROWS * ENC;
  float* hnf = m2z + ROWS * ENC;        // hn f32 (this step only)
  _Float16* h16A = (_Float16*)(hnf + ROWS * HID);   // f16 hidden, double buffered
  _Float16* h16B = h16A + ROWS * HID;
  _Float16* zm_h = h16B + ROWS * HID;   // [2048,256] = [z | m] f16
  _Float16* q_h  = zm_h + ROWS * 2 * ENC;
  _Float16* kk_h = q_h  + ROWS * QD;
  _Float16* WpEnc = kk_h + ROWS * QD;   // packed f16 weights
  _Float16* WpM1  = WpEnc + HID * ENC;
  _Float16* WpM2  = WpM1  + ENC * ENC;
  _Float16* WpU   = WpM2  + ENC * ENC;
  _Float16* WpQ   = WpU   + 2 * ENC * HID;
  _Float16* WpK   = WpQ   + HID * QD;

  float* outY = (float*)d_out;                            // [L,B]
  float* outA = outY + LSTEPS * BATCH;                    // [L,B,K,K]
  float* outM = outA + (size_t)LSTEPS * BATCH * KN * KN;  // [L,B,K]

  // ---- one-time per launch: zero h0, pack weights to f16 fragment order ----
  hipMemsetAsync(h16A, 0, ROWS * HID * sizeof(_Float16), stream);
  pack_w_kernel<<<4, 256, 0, stream>>>(Wenc + 2 * ENC, WpEnc, HID, ENC);
  pack_w_kernel<<<4, 256, 0, stream>>>(M1w, WpM1, ENC, ENC);
  pack_w_kernel<<<4, 256, 0, stream>>>(M2w, WpM2, ENC, ENC);
  pack_w_kernel<<<8, 256, 0, stream>>>(Uw,  WpU,  2 * ENC, HID);
  pack_w_kernel<<<2, 256, 0, stream>>>(Wq,  WpQ,  HID, QD);
  pack_w_kernel<<<2, 256, 0, stream>>>(Wk,  WpK,  HID, QD);

  _Float16* hcur = h16A; _Float16* hnext = h16B;
  for (int t = 0; t < LSTEPS; ++t) {
    const float* Et = E + (size_t)t * BATCH * KN * 2;
    const unsigned char* Pt = P + (size_t)t * BATCH * KN * KN;

    encode_kernel<<<256, 128, 0, stream>>>(Et, hcur, WpEnc, Wenc, benc, z, zm_h);
    gemm_h_kernel<<<256, 128, 0, stream>>>(zm_h, 2 * ENC, ENC, WpM1, M1b,
                                           m1z, nullptr, ENC, 0);
    gemm_h_kernel<<<256, 128, 0, stream>>>(zm_h, 2 * ENC, ENC, WpM2, M2b,
                                           m2z, nullptr, ENC, 0);
    message_kernel<<<256, 256, 0, stream>>>(Pt, m1z, m2z, zm_h);
    gemm_h_kernel<<<256, 128, 0, stream>>>(zm_h, 2 * ENC, 2 * ENC, WpU, Ub,
                                           hnf, hnext, HID, 1);
    gemm_h_kernel<<<128, 128, 0, stream>>>(hnext, HID, HID, WpQ, bq,
                                           nullptr, q_h, QD, 0);
    gemm_h_kernel<<<128, 128, 0, stream>>>(hnext, HID, HID, WpK, bk,
                                           nullptr, kk_h, QD, 0);
    attn_kernel<<<512, 128, 0, stream>>>(q_h, kk_h, outA + (size_t)t * BATCH * KN * KN);
    reduce_y_kernel<<<BATCH, 128, 0, stream>>>(z, hnf, decw, decb, outY + t * BATCH);
    mask_kernel<<<256, 256, 0, stream>>>(z, hnf, mskw, mskb, outM + (size_t)t * BATCH * KN);

    _Float16* tmp = hcur; hcur = hnext; hnext = tmp;
  }
}
